// CalibratedMultiHeadAttention_67319317397733
// MI455X (gfx1250) — matrix operand order repaired and checked by reading, not compiled, not run on hardware
//
#include <hip/hip_runtime.h>
#include <math.h>

// Problem constants
#define CB 4
#define CL 1024
#define CD 512
#define CH 8
#define CDK 64

typedef __attribute__((ext_vector_type(16))) __bf16 v16bf;
typedef __attribute__((ext_vector_type(8)))  __bf16 v8bf;
typedef __attribute__((ext_vector_type(8)))  float  v8f;

__device__ __forceinline__ v8f wmma_bf16(v16bf a, v16bf b, v8f c) {
  return __builtin_amdgcn_wmma_f32_16x16x32_bf16(false, a, false, b, (short)0, c, false, false);
}

__device__ __forceinline__ v16bf combine16(v8bf lo, v8bf hi) {
  v16bf a;
#pragma unroll
  for (int j = 0; j < 8; ++j) { a[j] = lo[j]; a[8 + j] = hi[j]; }
  return a;
}

// ---- Fragment loaders -------------------------------------------------------
// A 16x32 bf16 fragment: lane -> row lane&15; elem e -> k=(e<8?e:e+8)+8*(lane>>4)
__device__ __forceinline__ v16bf load_a_bf16(const __bf16* __restrict__ base, int lda,
                                             int lane, int k0) {
  const __bf16* r = base + (size_t)(lane & 15) * lda + k0 + ((lane >> 4) << 3);
  return combine16(*(const v8bf*)r, *(const v8bf*)(r + 16));
}

// B 32x16 bf16 fragment: lane -> col lane&15; elem e -> k=(e<8?e:e+8)+8*(lane>>4)
__device__ __forceinline__ v16bf load_b_bf16(const __bf16* __restrict__ B, int ldb,
                                             int k0, int n0, int lane) {
  const __bf16* r = B + (size_t)(k0 + ((lane >> 4) << 3)) * ldb + n0 + (lane & 15);
  v16bf b;
#pragma unroll
  for (int j = 0; j < 8; ++j) { b[j] = r[(size_t)j * ldb]; b[8 + j] = r[(size_t)(16 + j) * ldb]; }
  return b;
}

// ---- fp32 -> bf16 bulk convert ---------------------------------------------
__global__ void cvt_bf16(const float* __restrict__ src, __bf16* __restrict__ dst, int n2) {
  int i = blockIdx.x * blockDim.x + threadIdx.x;   // one thread = 2 elements
  if (i >= n2) return;
  float2 v = ((const float2*)src)[i];
  __bf16 a = (__bf16)v.x, b = (__bf16)v.y;
  unsigned p = ((unsigned)__builtin_bit_cast(unsigned short, b) << 16) |
               (unsigned)__builtin_bit_cast(unsigned short, a);
  ((unsigned*)dst)[i] = p;
}

// ---- Batched GEMM: C = act(alpha*(A@B) + bias), all-bf16 operands ----------
// CMODE: 0=C fp32, 1=C bf16, 2=C bf16 transposed.  ACT: 1 = sigmoid.
// 128 threads = 4 waves, each wave a 16x16 tile (block 32x32).
template <int CMODE, int ACT>
__global__ void gemm_wmma(const __bf16* __restrict__ Ap, const __bf16* __restrict__ Bp,
                          const float* __restrict__ bias, void* __restrict__ Cp,
                          int K, int lda, int ldb, int ldc,
                          long long sA, long long sB, long long sC, float alpha) {
  const int lane = threadIdx.x & 31;
  const int wave = threadIdx.x >> 5;
  const int bm = blockIdx.x * 32 + (wave >> 1) * 16;
  const int bn = blockIdx.y * 32 + (wave & 1) * 16;
  const int bz = blockIdx.z;
  const __bf16* Bb = Bp + (size_t)bz * sB;
  const __bf16* Ab = Ap + (size_t)bz * sA + (size_t)bm * lda;

  v8f acc = {0.f, 0.f, 0.f, 0.f, 0.f, 0.f, 0.f, 0.f};
  for (int k0 = 0; k0 < K; k0 += 32) {
    if (k0 + 32 < K)
      __builtin_prefetch(Ab + (size_t)(lane & 15) * lda + k0 + 32, 0, 0);
    acc = wmma_bf16(load_a_bf16(Ab, lda, lane, k0),
                    load_b_bf16(Bb, ldb, k0, bn, lane), acc);
  }

  const int n = lane & 15, half = lane >> 4;
  const float bv = bias ? bias[bn + n] : 0.f;
  if constexpr (CMODE == 2) {
    // transposed bf16 store: row = bn+n, cols bm+8*half.. contiguous -> one 16B store
    __bf16* Cb = (__bf16*)Cp + (size_t)bz * sC + (size_t)(bn + n) * ldc + bm + 8 * half;
    v8bf pk;
#pragma unroll
    for (int i = 0; i < 8; ++i) {
      float c = acc[i] * alpha + bv;
      if (ACT) c = 1.f / (1.f + __expf(-c));
      pk[i] = (__bf16)c;
    }
    *(v8bf*)Cb = pk;
  } else {
#pragma unroll
    for (int i = 0; i < 8; ++i) {
      const int m = i + 8 * half;
      float c = acc[i] * alpha + bv;
      if (ACT) c = 1.f / (1.f + __expf(-c));
      if constexpr (CMODE == 0)
        ((float*)Cp)[(size_t)bz * sC + (size_t)(bm + m) * ldc + bn + n] = c;
      else
        ((__bf16*)Cp)[(size_t)bz * sC + (size_t)(bm + m) * ldc + bn + n] = (__bf16)c;
    }
  }
}

// g = sigmoid(query @ gate_w + gate_b), one thread per row
__global__ void gate_kernel(const float* __restrict__ query, const float* __restrict__ gw,
                            const float* __restrict__ gb, float* __restrict__ g) {
  const int r = blockIdx.x * blockDim.x + threadIdx.x;
  if (r >= CB * CL) return;
  const float* qr = query + (size_t)r * CD;
  float s = gb[0];
  for (int d = 0; d < CD; ++d) s += qr[d] * gw[d];
  g[r] = 1.f / (1.f + __expf(-s));
}

// Fused attention: one wave per (b, h, 16-query tile); 3 waves per block.
// LDS per wave: 16x1024 fp32 scores + 16x1024 bf16 P + 16 sums.
#define WSTRIDE (16 * CL + 8192 + 16)   // in floats: 16384 + 8192 + 16
__global__ void attn_fused(const __bf16* __restrict__ q, const __bf16* __restrict__ kT,
                           const __bf16* __restrict__ v, const float* __restrict__ mbuf,
                           const float* __restrict__ g, __bf16* __restrict__ outh) {
  extern __shared__ float lds[];
  const int lane = threadIdx.x & 31;
  const int wave = threadIdx.x >> 5;
  const int idx = blockIdx.x * 3 + wave;       // 0..2048 (tail guarded)
  if (idx >= CB * CH * (CL / 16)) return;      // wave-uniform: EXEC stays all-ones
  const int qt = idx & 63;                     // query tile (L/16)
  const int h  = (idx >> 6) & 7;
  const int b  = idx >> 9;

  float*  srow = lds + (size_t)wave * WSTRIDE;        // 16x1024 fp32
  __bf16* pbuf = (__bf16*)(srow + 16 * CL);           // 16x1024 bf16
  float*  sums = (float*)(pbuf + 16 * CL);            // 16 fp32

  const __bf16* Qb  = q  + ((size_t)b * CL + qt * 16) * CD + h * CDK;   // rows, lda=CD
  const __bf16* KTb = kT + (size_t)b * CD * CL + (size_t)h * CDK * CL;  // rows=d, ld=CL
  const __bf16* Vb  = v  + (size_t)b * CL * CD + h * CDK;               // rows=l, ld=CD

  // Phase 1: S = Q @ K^T / sqrt(DK) into LDS (16 x 1024)
  for (int kt = 0; kt < 64; ++kt) {
    v8f acc = {0.f, 0.f, 0.f, 0.f, 0.f, 0.f, 0.f, 0.f};
    acc = wmma_bf16(load_a_bf16(Qb, CD, lane, 0),
                    load_b_bf16(KTb, CL, 0,  kt * 16, lane), acc);
    acc = wmma_bf16(load_a_bf16(Qb, CD, lane, 32),
                    load_b_bf16(KTb, CL, 32, kt * 16, lane), acc);
    const int n = lane & 15, half = lane >> 4;
#pragma unroll
    for (int i = 0; i < 8; ++i)
      srow[(size_t)(i + 8 * half) * CL + kt * 16 + n] = acc[i] * 0.125f;
  }

  // Phase 2: two lanes per row (contiguous halves), partials merged via shfl_xor.
  {
    const int r = lane & 15;
    const int piece = lane >> 4;                  // 0: j in [0,512), 1: [512,1024)
    float* row   = srow + (size_t)r * CL;
    float* half_ = row + piece * (CL / 2);
    const int qrow = qt * 16 + r;

    float mx = -3.4e38f;
    for (int t = 0; t < CL / 2; ++t) mx = fmaxf(mx, half_[t]);
    mx = fmaxf(mx, __shfl_xor(mx, 16, 32));

    float s1 = 0.f;
    for (int t = 0; t < CL / 2; ++t) {
      float p = __expf(half_[t] - mx); half_[t] = p; s1 += p;
    }
    s1 += __shfl_xor(s1, 16, 32);
    const float inv1 = 1.f / s1;

    const float gr = g[b * CL + qrow];
    const float* mhalf = mbuf + ((size_t)b * CL + qrow) * CL + piece * (CL / 2);
    float mx2 = -3.4e38f;
    for (int t = 0; t < CL / 2; ++t) {
      float p = half_[t] * inv1;                                  // p_attn
      float c = p * (gr + (1.f - gr) * __expf(1.f - mhalf[t]));   // attn_comb
      half_[t] = c; mx2 = fmaxf(mx2, c);
    }
    mx2 = fmaxf(mx2, __shfl_xor(mx2, 16, 32));

    __bf16* phalf = pbuf + (size_t)r * CL + piece * (CL / 2);
    float s2 = 0.f;
    for (int t = 0; t < CL / 2; ++t) {
      float e = __expf(half_[t] - mx2); s2 += e;
      phalf[t] = (__bf16)e;                       // separate buffer: race-free
    }
    s2 += __shfl_xor(s2, 16, 32);
    if (piece == 0) sums[r] = s2;
  }

  // Phase 3: out = P @ V (P bf16 in LDS, V bf16 global; direct b128 loads)
  v8f acc[4];
#pragma unroll
  for (int nb = 0; nb < 4; ++nb) acc[nb] = (v8f){0.f,0.f,0.f,0.f,0.f,0.f,0.f,0.f};
  for (int kt = 0; kt < 32; ++kt) {
    const int k0 = kt * 32;
    const __bf16* prow = pbuf + (size_t)(lane & 15) * CL + k0 + ((lane >> 4) << 3);
    v16bf a = combine16(*(const v8bf*)prow, *(const v8bf*)(prow + 16));
#pragma unroll
    for (int nb = 0; nb < 4; ++nb)
      acc[nb] = wmma_bf16(a, load_b_bf16(Vb, CD, k0, nb * 16, lane), acc[nb]);
  }
  const int n = lane & 15, half = lane >> 4;
#pragma unroll
  for (int nb = 0; nb < 4; ++nb)
#pragma unroll
    for (int i = 0; i < 8; ++i) {
      const int m = i + 8 * half;
      outh[((size_t)b * CL + qt * 16 + m) * CD + h * CDK + nb * 16 + n] =
          (__bf16)(acc[nb][i] / sums[m]);
    }
}

extern "C" void kernel_launch(void* const* d_in, const int* in_sizes, int n_in,
                              void* d_out, int out_size, void* d_ws, size_t ws_size,
                              hipStream_t stream) {
  (void)in_sizes; (void)n_in; (void)out_size; (void)ws_size;
  const float* query   = (const float*)d_in[0];
  const float* key_    = (const float*)d_in[1];
  const float* value   = (const float*)d_in[2];
  const float* wq_w    = (const float*)d_in[3];
  const float* wq_b    = (const float*)d_in[4];
  const float* wk_w    = (const float*)d_in[5];
  const float* wk_b    = (const float*)d_in[6];
  const float* wv_w    = (const float*)d_in[7];
  const float* wv_b    = (const float*)d_in[8];
  const float* dense_w = (const float*)d_in[9];
  const float* dense_b = (const float*)d_in[10];
  const float* gate_w  = (const float*)d_in[11];
  const float* gate_b  = (const float*)d_in[12];
  const float* mp_wq_w = (const float*)d_in[13];
  const float* mp_wq_b = (const float*)d_in[14];
  const float* mp_wk_w = (const float*)d_in[15];
  const float* mp_wk_b = (const float*)d_in[16];

  float* out  = (float*)d_out;                       // (B,L,D) fp32
  float* mout = out + (size_t)CB * CL * CD;          // (B,L,L) fp32

  // Workspace layout (bf16 activations & weights, fp32 gate)
  const size_t NT = (size_t)CB * CL * CD;            // 2,097,152 elements
  const size_t NW = (size_t)CD * CD;                 // 262,144 elements
  __bf16* qx   = (__bf16*)d_ws;                      // bf16 copies of inputs
  __bf16* kx   = qx  + NT;
  __bf16* vx   = kx  + NT;
  __bf16* qh   = vx  + NT;                           // (B,L,D)
  __bf16* khT  = qh  + NT;                           // (B,D,L)
  __bf16* vh   = khT + NT;                           // (B,L,D)
  __bf16* qpp  = vh  + NT;                           // (B,L,D)
  __bf16* kppT = qpp + NT;                           // (B,D,L)
  __bf16* oh   = kppT + NT;                          // (B,L,D)
  __bf16* w_q  = oh  + NT;
  __bf16* w_k  = w_q + NW;
  __bf16* w_v  = w_k + NW;
  __bf16* w_d  = w_v + NW;
  __bf16* w_mq = w_d + NW;
  __bf16* w_mk = w_mq + NW;
  float*  gbf  = (float*)(w_mk + NW);                // (B*L)

  // 1) fp32 -> bf16 once: inputs + weights
  {
    dim3 cb(256);
    dim3 cgT((unsigned)((NT / 2 + 255) / 256));
    cvt_bf16<<<cgT, cb, 0, stream>>>(query, qx, (int)(NT / 2));
    cvt_bf16<<<cgT, cb, 0, stream>>>(key_,  kx, (int)(NT / 2));
    cvt_bf16<<<cgT, cb, 0, stream>>>(value, vx, (int)(NT / 2));
    dim3 cgW((unsigned)((NW / 2 + 255) / 256));
    cvt_bf16<<<cgW, cb, 0, stream>>>(wq_w,    w_q,  (int)(NW / 2));
    cvt_bf16<<<cgW, cb, 0, stream>>>(wk_w,    w_k,  (int)(NW / 2));
    cvt_bf16<<<cgW, cb, 0, stream>>>(wv_w,    w_v,  (int)(NW / 2));
    cvt_bf16<<<cgW, cb, 0, stream>>>(dense_w, w_d,  (int)(NW / 2));
    cvt_bf16<<<cgW, cb, 0, stream>>>(mp_wq_w, w_mq, (int)(NW / 2));
    cvt_bf16<<<cgW, cb, 0, stream>>>(mp_wk_w, w_mk, (int)(NW / 2));
  }

  dim3 blk(128);
  const long long sAct = (long long)CL * CD;   // per-batch activation stride
  const long long sTr  = (long long)CD * CL;   // per-batch transposed stride

  // 2) Projections: all-bf16 operands, batched over B=4 (M=1024, N=512)
  dim3 gProj(CL / 32, CD / 32, CB);
  gemm_wmma<1,0><<<gProj, blk, 0, stream>>>(qx, w_q,  wq_b,    qh,
      CD, CD, CD, CD, sAct, 0, sAct, 1.f);
  gemm_wmma<2,0><<<gProj, blk, 0, stream>>>(kx, w_k,  wk_b,    khT,
      CD, CD, CD, CL, sAct, 0, sTr,  1.f);               // transposed (B,D,L)
  gemm_wmma<1,0><<<gProj, blk, 0, stream>>>(vx, w_v,  wv_b,    vh,
      CD, CD, CD, CD, sAct, 0, sAct, 1.f);
  gemm_wmma<1,0><<<gProj, blk, 0, stream>>>(qx, w_mq, mp_wq_b, qpp,
      CD, CD, CD, CD, sAct, 0, sAct, 1.f);
  gemm_wmma<2,0><<<gProj, blk, 0, stream>>>(kx, w_mk, mp_wk_b, kppT,
      CD, CD, CD, CL, sAct, 0, sTr,  1.f);               // transposed (B,D,L)

  // 3) Gate
  gate_kernel<<<(CB * CL + 255) / 256, 256, 0, stream>>>(query, gate_w, gate_b, gbf);

  // 4) m = sigmoid(qp @ kp^T / sqrt(D)) -> straight into d_out's m region (fp32)
  dim3 gM(CL / 32, CL / 32, CB);
  gemm_wmma<0,1><<<gM, blk, 0, stream>>>(qpp, kppT, nullptr, mout,
      CD, CD, CL, CL, sAct, sTr, (long long)CL * CL,
      0.04419417382415922f /* 1/sqrt(512) */);

  // 5) Fused attention: 2048 wave-tiles, 3 waves/block, ~288KB dynamic LDS per WG
  const size_t shmem = 3 * (size_t)WSTRIDE * sizeof(float);
  hipFuncSetAttribute(reinterpret_cast<const void*>(attn_fused),
                      hipFuncAttributeMaxDynamicSharedMemorySize, (int)shmem);
  const int nTiles = CB * CH * (CL / 16);              // 2048
  attn_fused<<<(nTiles + 2) / 3, 96, shmem, stream>>>(qh, khT, vh, mout, gbf, oh);

  // 6) Final dense projection: A bf16, C fp32 into d_out (M=4096 flat)
  dim3 gD(CB * CL / 32, CD / 32, 1);
  gemm_wmma<0,0><<<gD, blk, 0, stream>>>(oh, w_d, dense_b, out,
      CD, CD, CD, CD, 0, 0, 0, 1.f);
}
